// SNNModel_90426241450244
// MI455X (gfx1250) — 16-bit and 8-bit matrix operand order repaired and checked by reading, compiled, not run on hardware
//
#include <hip/hip_runtime.h>


typedef __bf16 bf16_t;
typedef __attribute__((ext_vector_type(16))) __bf16 v16bf;
typedef __attribute__((ext_vector_type(8)))  __bf16 v8bf;
typedef __attribute__((ext_vector_type(8)))  float  v8f;
typedef __attribute__((ext_vector_type(8)))  int    v8i;
typedef __attribute__((ext_vector_type(4)))  int    v4i;
typedef __attribute__((ext_vector_type(2)))  int    v2i;

#define T_STEPS   200
#define BATCH     256
#define IN_SIZE   784
#define NN        1024
#define OUT_SZ    10
#define K_IN_PAD  800

#define V_REST    (-65.0f)
#define THRESH0   (-50.0f)
#define BETA      5.0f
#define INV_TAU_M 0.05f
#define DT_TAUTH  0.01f


static __device__ __forceinline__ v8f wmma_bf16(v16bf a, v16bf b, v8f c) {
  return __builtin_amdgcn_wmma_f32_16x16x32_bf16(false, a, false, b,
                                                 (short)0, c, false, false);
}
static __device__ __forceinline__ v8f wmma_fp8(v8i a, v8i b, v8f c) {
  return __builtin_amdgcn_wmma_f32_16x16x64_fp8_fp8(a, b, (short)0, c,
                                                    false, false);
}

static __device__ __forceinline__ void async_copy_b128(unsigned lds_off,
                                                       const void* gptr) {
  unsigned long long ga = (unsigned long long)gptr;
  asm volatile("global_load_async_to_lds_b128 %0, %1, off"
               :: "v"(lds_off), "v"(ga) : "memory");
}
static __device__ __forceinline__ void wait_async0() {
  asm volatile("s_wait_asynccnt 0" ::: "memory");
}

static __device__ __forceinline__ unsigned char f32_to_fp8(float f) {
  unsigned int u = __float_as_uint(f);
  unsigned char s = (unsigned char)((u >> 24) & 0x80u);
  int exp = (int)((u >> 23) & 0xFF) - 127;
  unsigned int mant = u & 0x7FFFFFu;
  if (((u >> 23) & 0xFF) == 0xFF) return (unsigned char)(s | 0x7E);
  if (exp >= -6) {
    unsigned int m = mant >> 20;
    unsigned int rem = mant & 0xFFFFFu;
    if (rem > 0x80000u || (rem == 0x80000u && (m & 1u))) ++m;
    if (m == 8u) { m = 0u; ++exp; }
    if (exp > 8) return (unsigned char)(s | 0x7E);
    return (unsigned char)(s | (unsigned char)((exp + 7) << 3) | (unsigned char)m);
  }
  float a = __uint_as_float(u & 0x7FFFFFFFu);
  int q = (int)rintf(a * 512.0f);
  if (q > 7) q = 7;
  return (unsigned char)(s | (unsigned char)q);
}

static __device__ __forceinline__ v16bf load_a_bf16_lds(const bf16_t* base,
                                                        int lane, int srow) {
  int m  = lane & 15;
  int kh = (lane >> 4) << 4;
  v8bf lo = *(const v8bf*)(base + m * srow + kh);
  v8bf hi = *(const v8bf*)(base + m * srow + 8 + kh);
  return __builtin_shufflevector(lo, hi, 0, 1, 2, 3, 4, 5, 6, 7,
                                 8, 9, 10, 11, 12, 13, 14, 15);
}

static __device__ __forceinline__ v16bf load_b_bf16(const bf16_t* wt, int n0,
                                                    int k0, int lane, int srow) {
  int n  = n0 + (lane & 15);
  int kk = k0 + ((lane >> 4) << 4);
  return *(const v16bf*)(wt + (size_t)n * srow + kk);
}

static __device__ __forceinline__ v8i load_a_fp8_lds(const unsigned char* base,
                                                     int lane, int srow, int k0) {
  int m  = lane & 15;
  int kh = (lane >> 4) << 4;
  const unsigned char* r = base + m * srow + k0 + kh;
  v2i c0 = *(const v2i*)(r);
  v2i c1 = *(const v2i*)(r + 8);
  v2i c2 = *(const v2i*)(r + 32);
  v2i c3 = *(const v2i*)(r + 40);
  v8i a;
  a[0] = c0[0]; a[1] = c0[1]; a[2] = c1[0]; a[3] = c1[1];
  a[4] = c2[0]; a[5] = c2[1]; a[6] = c3[0]; a[7] = c3[1];
  return a;
}

static __device__ __forceinline__ v8i load_b_fp8(const unsigned char* wt,
                                                 int n0, int k0, int lane) {
  int n  = n0 + (lane & 15);
  int kk = k0 + ((lane >> 4) << 4);
  const unsigned char* r = wt + (size_t)n * NN + kk;
  v4i c0 = *(const v4i*)(r);
  v4i c1 = *(const v4i*)(r + 32);
  v8i b;
  b[0] = c0[0]; b[1] = c0[1]; b[2] = c0[2]; b[3] = c0[3];
  b[4] = c1[0]; b[5] = c1[1]; b[6] = c1[2]; b[7] = c1[3];
  return b;
}


__global__ void prep_wrec_fp8(const float* __restrict__ wrec,
                              unsigned char* __restrict__ wrecT) {
  int idx = blockIdx.x * blockDim.x + threadIdx.x;
  int j = idx >> 10;
  int i = idx & (NN - 1);
  wrecT[idx] = f32_to_fp8(wrec[i * NN + j]);
}

__global__ void prep_win_bf16(const float* __restrict__ win,
                              bf16_t* __restrict__ wth,
                              bf16_t* __restrict__ wtl) {
  int idx = blockIdx.x * blockDim.x + threadIdx.x;
  int j = idx / K_IN_PAD;
  int k = idx - j * K_IN_PAD;
  float v = (k < IN_SIZE) ? win[k * NN + j] : 0.0f;
  bf16_t h = (bf16_t)v;
  wth[idx] = h;
  wtl[idx] = (bf16_t)(v - (float)h);
}


__global__ __launch_bounds__(256) void snn_phase1(
    const float*  __restrict__ xs,
    const bf16_t* __restrict__ wth,
    const bf16_t* __restrict__ wtl,
    float*        __restrict__ net1)
{
  const int SROW = 808;
  __shared__ bf16_t Ah[16 * SROW];
  __shared__ bf16_t Al[16 * SROW];

  size_t row0 = (size_t)blockIdx.x * 16;

  for (int idx = threadIdx.x; idx < 16 * K_IN_PAD; idx += 256) {
    int m = idx / K_IN_PAD;
    int k = idx - m * K_IN_PAD;
    float v = (k < IN_SIZE) ? xs[(row0 + m) * IN_SIZE + k] : 0.0f;
    bf16_t h = (bf16_t)v;
    Ah[m * SROW + k] = h;
    Al[m * SROW + k] = (bf16_t)(v - (float)h);
  }
  __syncthreads();

  int wave = threadIdx.x >> 5;
  int lane = threadIdx.x & 31;
  int ncol = wave * 128;

  v8f C[8];
#pragma unroll
  for (int nt = 0; nt < 8; ++nt) C[nt] = (v8f){0, 0, 0, 0, 0, 0, 0, 0};

  for (int ks = 0; ks < K_IN_PAD / 32; ++ks) {
    int k0 = ks * 32;
    v16bf ah = load_a_bf16_lds(Ah + k0, lane, SROW);
    v16bf al = load_a_bf16_lds(Al + k0, lane, SROW);
#pragma unroll
    for (int nt = 0; nt < 8; ++nt) {
      int n0 = ncol + nt * 16;
      v16bf bh = load_b_bf16(wth, n0, k0, lane, K_IN_PAD);
      v16bf bl = load_b_bf16(wtl, n0, k0, lane, K_IN_PAD);
      C[nt] = wmma_bf16(ah, bh, C[nt]);
      C[nt] = wmma_bf16(ah, bl, C[nt]);
      C[nt] = wmma_bf16(al, bh, C[nt]);
    }
  }

  float* outp = net1 + row0 * NN;
  int n15 = lane & 15, mb = (lane >> 4) << 3;
#pragma unroll
  for (int nt = 0; nt < 8; ++nt) {
    int n = ncol + nt * 16 + n15;
#pragma unroll
    for (int r = 0; r < 8; ++r)
      outp[(size_t)(mb + r) * NN + n] = C[nt][r];
  }
}


#define NSROW 1036
#define SSROW 1040

__global__ __launch_bounds__(256) void snn_phase2(
    const unsigned char* __restrict__ wrecT,
    float*               __restrict__ hidden,
    float*               __restrict__ counts)
{
  __shared__ float         Nbuf[2 * 16 * NSROW];
  __shared__ unsigned char Sbuf[16 * SSROW];

  int lane = threadIdx.x & 31;
  int b0   = blockIdx.x * 16;
  int ncol = (threadIdx.x >> 5) * 128;
  int n15  = lane & 15, mb = (lane >> 4) << 3;

  unsigned nbuf_base = (unsigned)(unsigned long long)(const void*)&Nbuf[0];

  for (int i = threadIdx.x; i < 16 * SSROW; i += 256) Sbuf[i] = 0;

  {
    const unsigned char* src =
        (const unsigned char*)(hidden + (size_t)b0 * NN);
    unsigned dst = nbuf_base;
    for (int c = threadIdx.x; c < 16 * 256; c += 256) {
      int m = c >> 8, i = c & 255;
      async_copy_b128(dst + m * (NSROW * 4) + i * 16, src + m * 4096 + i * 16);
    }
  }
  wait_async0();
  __syncthreads();

  float mp[8][8], th[8][8], cnt[8][8];
#pragma unroll
  for (int nt = 0; nt < 8; ++nt)
#pragma unroll
    for (int r = 0; r < 8; ++r) {
      mp[nt][r] = V_REST; th[nt][r] = THRESH0; cnt[nt][r] = 0.0f;
    }

  for (int t = 0; t < T_STEPS; ++t) {
    float* hid_t = hidden + (size_t)t * BATCH * NN + (size_t)b0 * NN;

    if (t + 1 < T_STEPS) {
      const unsigned char* src = (const unsigned char*)(hid_t + (size_t)BATCH * NN);
      unsigned dst = nbuf_base + (unsigned)(((t + 1) & 1) * 16 * NSROW * 4);
      for (int c = threadIdx.x; c < 16 * 256; c += 256) {
        int m = c >> 8, i = c & 255;
        async_copy_b128(dst + m * (NSROW * 4) + i * 16, src + m * 4096 + i * 16);
      }
    }

    const float* Nb = &Nbuf[(t & 1) * 16 * NSROW];
    v8f C[8];
#pragma unroll
    for (int nt = 0; nt < 8; ++nt) {
      int col = ncol + nt * 16 + n15;
#pragma unroll
      for (int r = 0; r < 8; ++r)
        C[nt][r] = Nb[(mb + r) * NSROW + col];
    }

    for (int ks = 0; ks < NN / 64; ++ks) {
      int k0 = ks * 64;
      v8i a = load_a_fp8_lds(Sbuf, lane, SSROW, k0);
#pragma unroll
      for (int nt = 0; nt < 8; ++nt) {
        v8i b = load_b_fp8(wrecT, ncol + nt * 16, k0, lane);
        C[nt] = wmma_fp8(a, b, C[nt]);
      }
    }
    __syncthreads();

#pragma unroll
    for (int nt = 0; nt < 8; ++nt) {
      int col = ncol + nt * 16 + n15;
#pragma unroll
      for (int r = 0; r < 8; ++r) {
        float m = mp[nt][r];
        m = m + (V_REST - m) * INV_TAU_M + C[nt][r];
        float spike = (m >= th[nt][r]) ? 1.0f : 0.0f;
        th[nt][r] = th[nt][r] + BETA * spike
                  - (th[nt][r] - THRESH0) * DT_TAUTH;
        mp[nt][r] = (spike > 0.0f) ? V_REST : m;
        cnt[nt][r] += spike;
        hid_t[(size_t)(mb + r) * NN + col] = spike;
        Sbuf[(mb + r) * SSROW + col] =
            (spike > 0.0f) ? (unsigned char)0x38 : (unsigned char)0x00;
      }
    }

    wait_async0();
    __syncthreads();
  }

#pragma unroll
  for (int nt = 0; nt < 8; ++nt) {
    int col = ncol + nt * 16 + n15;
#pragma unroll
    for (int r = 0; r < 8; ++r)
      counts[(size_t)(b0 + mb + r) * NN + col] = cnt[nt][r];
  }
}


__global__ void snn_logits(const float* __restrict__ counts,
                           const float* __restrict__ wout,
                           float* __restrict__ logits) {
  int b = blockIdx.x;
  int o = threadIdx.x;
  if (o < OUT_SZ) {
    float acc = 0.0f;
    for (int i = 0; i < NN; ++i)
      acc += counts[b * NN + i] * wout[i * OUT_SZ + o];
    logits[b * OUT_SZ + o] = acc;
  }
}


extern "C" void kernel_launch(void* const* d_in, const int* in_sizes, int n_in,
                              void* d_out, int out_size, void* d_ws,
                              size_t ws_size, hipStream_t stream) {
  (void)in_sizes; (void)n_in; (void)out_size; (void)ws_size;

  const float* xs   = (const float*)d_in[0];
  const float* win  = (const float*)d_in[1];
  const float* wrec = (const float*)d_in[2];
  const float* wout = (const float*)d_in[3];

  float* out    = (float*)d_out;
  float* logits = out;
  float* hidden = out + BATCH * OUT_SZ;

  unsigned char* ws = (unsigned char*)d_ws;
  unsigned char* wrecT = ws;
  bf16_t* wth = (bf16_t*)(ws + (size_t)NN * NN);
  bf16_t* wtl = wth + (size_t)NN * K_IN_PAD;
  float* counts = (float*)(ws + (size_t)NN * NN
                              + (size_t)2 * NN * K_IN_PAD * sizeof(bf16_t));

  prep_wrec_fp8<<<(NN * NN) / 256, 256, 0, stream>>>(wrec, wrecT);
  prep_win_bf16<<<(NN * K_IN_PAD) / 256, 256, 0, stream>>>(win, wth, wtl);
  snn_phase1<<<(T_STEPS * BATCH) / 16, 256, 0, stream>>>(xs, wth, wtl, hidden);
  snn_phase2<<<BATCH / 16, 256, 0, stream>>>(wrecT, hidden, counts);
  snn_logits<<<BATCH, 32, 0, stream>>>(counts, wout, logits);
}
